// HTN_45226005627322
// MI455X (gfx1250) — matrix operand order repaired and checked by reading, not compiled, not run on hardware
//
#include <hip/hip_runtime.h>

#define NNODES 50000
#define NEDGES 400000
#define FIN    128
#define NH     4
#define FOUT   64
#define CH     (NH*FOUT)   // 256
#define AHID   64
#define K1     (3*FOUT)    // 192

typedef _Float16 v16h __attribute__((ext_vector_type(16)));
typedef _Float16 v8h  __attribute__((ext_vector_type(8)));
typedef float    v8f  __attribute__((ext_vector_type(8)));

union Frag { v16h v; v8h h[2]; };

// ---------------------------------------------------------------- utilities
__global__ void k_f32_to_f16(const float* __restrict__ in,
                             _Float16* __restrict__ out, int n) {
  int i = blockIdx.x * blockDim.x + threadIdx.x;
  if (i < n) out[i] = (_Float16)in[i];
}

__global__ void k_zero(float* __restrict__ p, int n) {
  int i = blockIdx.x * blockDim.x + threadIdx.x;
  if (i < n) p[i] = 0.f;
}

// Swizzle a row-major fp32 weight matrix W[K][Ncols] into WMMA B-fragment
// order (f16): frag f = t*(Ncols/16)+jc ; per lane 16 contiguous halves.
// B layout used here (contiguous k; A uses the same order): lane L holds column
// n = L%16, K = 32*t + 16*(L>=16) + q, q = 0..15 in order.
__global__ void k_swz_b(const float* __restrict__ W,
                        _Float16* __restrict__ out, int K, int Ncols) {
  int i = blockIdx.x * blockDim.x + threadIdx.x;
  if (i >= K * Ncols) return;
  int q    = i & 15;
  int lane = (i >> 4) & 31;
  int f    = i >> 9;
  int ntiles = Ncols >> 4;
  int t  = f / ntiles;
  int jc = f % ntiles;
  int k  = 32 * t + 16 * (lane >> 4) + q;
  int c  = 16 * jc + (lane & 15);
  out[i] = (_Float16)W[k * Ncols + c];
}

// ---------------------------------------------------------------- proj GEMM
// P[N,256] = X[N,K] @ W ; one wave computes a 16x64 tile.
// A layout (contiguous k, same order as B): lane L holds row M=L%16, halves q=0..15 map
// to K = 32t + 16*(L>=16) + q  -> one contiguous 16-half run.
template <int K>
__global__ void __launch_bounds__(32)
k_gemm16(const _Float16* __restrict__ X, const _Float16* __restrict__ Bswz,
         float* __restrict__ Pf32, _Float16* __restrict__ Pf16) {
  const int lane = threadIdx.x;
  const int ln = lane & 15, hi = lane >> 4;
  const int off0 = hi ? 8 : 0;
  const int bm = blockIdx.x;      // 16-row tile
  const int nt = blockIdx.y;      // 64-col block (0..3)
  const int row = bm * 16 + ln;

  const v8f vzero = {0.f,0.f,0.f,0.f,0.f,0.f,0.f,0.f};
  v8f acc[4];
#pragma unroll
  for (int j = 0; j < 4; ++j) acc[j] = vzero;

  const _Float16* xr = X + (size_t)row * K;
#pragma unroll
  for (int t = 0; t < K / 32; ++t) {
    Frag A;
    A.h[0] = *(const v8h*)(xr + 32 * t + 16 * hi);
    A.h[1] = *(const v8h*)(xr + 32 * t + 16 * hi + 8);
#pragma unroll
    for (int j = 0; j < 4; ++j) {
      const int f = t * 16 + (nt * 4 + j);   // Ncols/16 == 16
      const _Float16* bp = Bswz + ((size_t)f * 32 + lane) * 16;
      Frag B;
      B.h[0] = *(const v8h*)(bp);
      B.h[1] = *(const v8h*)(bp + 8);
      acc[j] = __builtin_amdgcn_wmma_f32_16x16x32_f16(
          false, A.v, false, B.v, (short)0, acc[j], false, false);
    }
  }
  // C layout: VGPR g -> row M = g + 8*hi, col N = ln
#pragma unroll
  for (int j = 0; j < 4; ++j)
#pragma unroll
    for (int g = 0; g < 8; ++g) {
      const int r = bm * 16 + g + 8 * hi;
      const int c = nt * 64 + j * 16 + ln;
      const float v = acc[j][g];
      Pf32[(size_t)r * CH + c] = v;
      Pf16[(size_t)r * CH + c] = (_Float16)v;
    }
}

// ---------------------------------------------------------------- edge kernel
// Per wave: 16 rows = 4 edges x 4 heads (head minor). comb = [pi|pj|pk] (192)
// -> 6 WMMA k-steps x 4 n-tiles, relu(+ab1), dot with aw2 (+ab2), leaky,
// softmax over heads (lane-local), alpha * pj * pk scattered with atomics.
__global__ void __launch_bounds__(256)
k_edge(const _Float16* __restrict__ Pf16, const float* __restrict__ Pf32,
       const int* __restrict__ eidx, const _Float16* __restrict__ Aw1swz,
       const float* __restrict__ ab1, const float* __restrict__ aw2,
       const float* __restrict__ ab2, float* __restrict__ agg) {
  __shared__ _Float16 sB[K1 * AHID];   // 12288 halves = 24 KB

  const int tid = threadIdx.x;
  for (int i = tid; i < (K1 * AHID) / 8; i += 256)
    ((v8h*)sB)[i] = ((const v8h*)Aw1swz)[i];
  __syncthreads();

  const int wave = tid >> 5, lane = tid & 31;
  const int ln = lane & 15, hi = lane >> 4;
  const int off0 = hi ? 8 : 0;
  const int e0 = (blockIdx.x * 8 + wave) * 4;

  // lane's A-row m = ln -> (edge, head)
  const int em = e0 + (ln >> 2);
  const int hm = ln & 3;
  const int ni = eidx[em];
  const int nj = eidx[NEDGES + em];
  const int nk = eidx[2 * NEDGES + em];

  const _Float16* seg0 = Pf16 + (size_t)ni * CH + hm * 64;
  const _Float16* seg1 = Pf16 + (size_t)nj * CH + hm * 64;
  const _Float16* seg2 = Pf16 + (size_t)nk * CH + hm * 64;

  const v8f vzero = {0.f,0.f,0.f,0.f,0.f,0.f,0.f,0.f};
  v8f acc[4];
#pragma unroll
  for (int j = 0; j < 4; ++j) acc[j] = vzero;

#pragma unroll
  for (int t = 0; t < 6; ++t) {
    const _Float16* base = (t < 2) ? seg0 : (t < 4) ? seg1 : seg2;
    const _Float16* p = base + (t & 1) * 32;
    Frag A;
    A.h[0] = *(const v8h*)(p + 16 * hi);
    A.h[1] = *(const v8h*)(p + 16 * hi + 8);
#pragma unroll
    for (int j = 0; j < 4; ++j) {
      const _Float16* bp = sB + ((t * 4 + j) * 32 + lane) * 16;
      Frag B;
      B.h[0] = *(const v8h*)bp;
      B.h[1] = *(const v8h*)(bp + 8);
      acc[j] = __builtin_amdgcn_wmma_f32_16x16x32_f16(
          false, A.v, false, B.v, (short)0, acc[j], false, false);
    }
  }

  // s = relu(h + ab1) . aw2  (per-lane partial over its 4 columns)
  float partial[8];
#pragma unroll
  for (int g = 0; g < 8; ++g) partial[g] = 0.f;
#pragma unroll
  for (int j = 0; j < 4; ++j) {
    const float b1 = ab1[j * 16 + ln];
    const float w2 = aw2[j * 16 + ln];
#pragma unroll
    for (int g = 0; g < 8; ++g) {
      float v = acc[j][g] + b1;
      v = v > 0.f ? v : 0.f;
      partial[g] += v * w2;
    }
  }
  const float b2 = ab2[0];
#pragma unroll
  for (int g = 0; g < 8; ++g) {   // reduce over the 16 lanes of each half
    float s = partial[g];
    s += __shfl_xor(s, 1);
    s += __shfl_xor(s, 2);
    s += __shfl_xor(s, 4);
    s += __shfl_xor(s, 8);
    s += b2;
    partial[g] = s >= 0.f ? s : 0.2f * s;   // LeakyReLU(0.2)
  }

  // softmax over heads: rows of one edge are 4 consecutive g's (lane-local)
  float alpha[8];
#pragma unroll
  for (int grp = 0; grp < 2; ++grp) {
    float m = partial[grp * 4];
#pragma unroll
    for (int u = 1; u < 4; ++u) m = fmaxf(m, partial[grp * 4 + u]);
    float ev[4], sum = 0.f;
#pragma unroll
    for (int u = 0; u < 4; ++u) {
      ev[u] = __expf(partial[grp * 4 + u] - m);
      sum += ev[u];
    }
    const float inv = 1.f / sum;
#pragma unroll
    for (int u = 0; u < 4; ++u) alpha[grp * 4 + u] = ev[u] * inv;
  }

  // scatter: agg[src] += alpha * pj * pk  (fp32, 2 cols/lane per row)
#pragma unroll
  for (int r = 0; r < 16; ++r) {
    const float a  = __shfl(alpha[r & 7], (r < 8) ? 0 : 16);
    const int srcn = __shfl(ni, r);
    const int n1   = __shfl(nj, r);
    const int n2   = __shfl(nk, r);
    const int hh   = r & 3;
    const float2 vj = ((const float2*)(Pf32 + (size_t)n1 * CH + hh * 64))[lane];
    const float2 vk = ((const float2*)(Pf32 + (size_t)n2 * CH + hh * 64))[lane];
    float* dst = agg + (size_t)srcn * CH + hh * 64 + 2 * lane;
    atomicAdd(dst,     a * vj.x * vk.x);
    atomicAdd(dst + 1, a * vj.y * vk.y);
  }
}

// ---------------------------------------------------------------- epilogues
__global__ void k_fin1(const float* __restrict__ P, const float* __restrict__ agg,
                       const float* __restrict__ theta, const float* __restrict__ bias,
                       _Float16* __restrict__ out16, int n) {
  int i = blockIdx.x * blockDim.x + threadIdx.x;
  if (i >= n) return;
  const int c = i & (CH - 1);
  float v = theta[c] * P[i] + agg[i] + bias[c];
  v = v > 0.f ? v : (__expf(v) - 1.f);   // ELU
  out16[i] = (_Float16)v;
}

__global__ void k_fin2(const float* __restrict__ P, const float* __restrict__ agg,
                       const float* __restrict__ theta, const float* __restrict__ bias,
                       float* __restrict__ out, int n) {
  int i = blockIdx.x * blockDim.x + threadIdx.x;
  if (i >= n) return;
  const int f = i & 63;
  const int node = i >> 6;
  float s = 0.f;
#pragma unroll
  for (int h = 0; h < 4; ++h) {
    const size_t idx = (size_t)node * CH + h * 64 + f;
    s += theta[h * 64 + f] * P[idx] + agg[idx];
  }
  out[i] = s * 0.25f + bias[f];          // mean over heads + bias
}

// ---------------------------------------------------------------- launcher
extern "C" void kernel_launch(void* const* d_in, const int* in_sizes, int n_in,
                              void* d_out, int out_size, void* d_ws, size_t ws_size,
                              hipStream_t stream) {
  const float* x      = (const float*)d_in[0];
  const int*   eidx   = (const int*)  d_in[1];
  const float* Wp1    = (const float*)d_in[2];
  const float* aw1_1  = (const float*)d_in[3];
  const float* ab1_1  = (const float*)d_in[4];
  const float* aw2_1  = (const float*)d_in[5];
  const float* ab2_1  = (const float*)d_in[6];
  const float* theta1 = (const float*)d_in[7];
  const float* bias1  = (const float*)d_in[8];
  const float* Wp2    = (const float*)d_in[9];
  const float* aw1_2  = (const float*)d_in[10];
  const float* ab1_2  = (const float*)d_in[11];
  const float* aw2_2  = (const float*)d_in[12];
  const float* ab2_2  = (const float*)d_in[13];
  const float* theta2 = (const float*)d_in[14];
  const float* bias2  = (const float*)d_in[15];

  char* ws = (char*)d_ws;
  _Float16* xf16    = (_Float16*)ws; ws += (size_t)NNODES * CH * 2;   // also out1_f16
  float*    projf32 = (float*)ws;    ws += (size_t)NNODES * CH * 4;
  _Float16* projf16 = (_Float16*)ws; ws += (size_t)NNODES * CH * 2;
  float*    agg     = (float*)ws;    ws += (size_t)NNODES * CH * 4;
  _Float16* wp1swz  = (_Float16*)ws; ws += (size_t)FIN * CH * 2;
  _Float16* wp2swz  = (_Float16*)ws; ws += (size_t)CH * CH * 2;
  _Float16* aw11swz = (_Float16*)ws; ws += (size_t)K1 * AHID * 2;
  _Float16* aw12swz = (_Float16*)ws; ws += (size_t)K1 * AHID * 2;

  // ---- prep: f16 conversion + fragment swizzles
  k_f32_to_f16<<<(NNODES * FIN + 255) / 256, 256, 0, stream>>>(x, xf16, NNODES * FIN);
  k_swz_b<<<(FIN * CH + 255) / 256, 256, 0, stream>>>(Wp1, wp1swz, FIN, CH);
  k_swz_b<<<(CH * CH + 255) / 256, 256, 0, stream>>>(Wp2, wp2swz, CH, CH);
  k_swz_b<<<(K1 * AHID + 255) / 256, 256, 0, stream>>>(aw1_1, aw11swz, K1, AHID);
  k_swz_b<<<(K1 * AHID + 255) / 256, 256, 0, stream>>>(aw1_2, aw12swz, K1, AHID);

  const dim3 ggemm(NNODES / 16, 4);

  // ---- layer 1
  k_gemm16<FIN><<<ggemm, 32, 0, stream>>>(xf16, wp1swz, projf32, projf16);
  k_zero<<<(NNODES * CH + 255) / 256, 256, 0, stream>>>(agg, NNODES * CH);
  k_edge<<<NEDGES / 32, 256, 0, stream>>>(projf16, projf32, eidx, aw11swz,
                                          ab1_1, aw2_1, ab2_1, agg);
  k_fin1<<<(NNODES * CH + 255) / 256, 256, 0, stream>>>(projf32, agg, theta1,
                                                        bias1, xf16, NNODES * CH);
  // ---- layer 2
  k_gemm16<CH><<<ggemm, 32, 0, stream>>>(xf16, wp2swz, projf32, projf16);
  k_zero<<<(NNODES * CH + 255) / 256, 256, 0, stream>>>(agg, NNODES * CH);
  k_edge<<<NEDGES / 32, 256, 0, stream>>>(projf16, projf32, eidx, aw12swz,
                                          ab1_2, aw2_2, ab2_2, agg);
  k_fin2<<<(NNODES * FOUT + 255) / 256, 256, 0, stream>>>(projf32, agg, theta2,
                                                          bias2, (float*)d_out,
                                                          NNODES * FOUT);
}
